// Decoder_51376398794985
// MI455X (gfx1250) — compile-verified, matrix operand order checked by reading, not run on hardware
//
#include <hip/hip_runtime.h>
#include <hip/hip_bf16.h>
#include <math.h>

// Problem constants (match reference)
#define BB 512
#define SS 128
#define TT 128
#define DD 512
#define VV 128
#define KK (3*DD)     // 1536 = [emb|ctx|h]
#define NG (4*DD)     // 2048 gate columns

typedef __bf16 bf16_t;
typedef bf16_t bf16x8  __attribute__((ext_vector_type(8)));
typedef bf16_t bf16x16 __attribute__((ext_vector_type(16)));
typedef float  v8f     __attribute__((ext_vector_type(8)));
typedef float  f32x4   __attribute__((ext_vector_type(4)));
typedef unsigned u32x4 __attribute__((ext_vector_type(4)));
typedef int      i32x8 __attribute__((ext_vector_type(8)));
typedef int      i32x4 __attribute__((ext_vector_type(4)));

// hardware f32->bf16 conversion (backend emits v_cvt for gfx1250)
__device__ inline bf16_t f2bf(float f) { return (bf16_t)f; }

__device__ inline float sigm(float x) { return 1.0f / (1.0f + expf(-x)); }

// A/B fragment loaders for V_WMMA_F32_16X16X32_BF16 (wave32).
// Caller passes p = row_base + k0 + half*8; lo 8 elems at +0, hi 8 at +16
// matching the 16-bit A-matrix 16x32 VGPR layout (ISA 7.12.2).
__device__ inline bf16x16 load_frag_bf16(const bf16_t* p) {
    bf16x8 lo = *(const bf16x8*)p;
    bf16x8 hi = *(const bf16x8*)(p + 16);
    bf16x16 r;
#pragma unroll
    for (int i = 0; i < 8; ++i) { r[i] = lo[i]; r[i + 8] = hi[i]; }
    return r;
}

__device__ inline bf16x16 load_frag_f32(const float* p) {
    f32x4 a0 = *(const f32x4*)p;
    f32x4 a1 = *(const f32x4*)(p + 4);
    f32x4 b0 = *(const f32x4*)(p + 16);
    f32x4 b1 = *(const f32x4*)(p + 20);
    bf16x16 r;
#pragma unroll
    for (int i = 0; i < 4; ++i) {
        r[i]      = f2bf(a0[i]);
        r[i + 4]  = f2bf(a1[i]);
        r[i + 8]  = f2bf(b0[i]);
        r[i + 12] = f2bf(b1[i]);
    }
    return r;
}

__device__ inline v8f wmma_bf16(bf16x16 a, bf16x16 b, v8f c) {
    return __builtin_amdgcn_wmma_f32_16x16x32_bf16(
        /*neg_a=*/false, a, /*neg_b=*/false, b,
        /*c_mod=*/(short)0, c, /*reuse_a=*/false, /*reuse_b=*/false);
}

// ---------------- one-time prep kernels ----------------

// WT[n*K + k] = bf16(W[k*N + n])   (store B transposed so B-frag loads are contiguous)
__global__ void transpose_to_bf16(const float* __restrict__ W, bf16_t* __restrict__ WT,
                                  int K, int N) {
    size_t tid = (size_t)blockIdx.x * blockDim.x + threadIdx.x;
    size_t total = (size_t)K * N;
    if (tid >= total) return;
    int n = (int)(tid / K), k = (int)(tid % K);
    WT[tid] = f2bf(W[(size_t)k * N + n]);
}

// Combined gate weight, transposed: WgT[n*1536 + k], k<1024 -> W_ih, else W_hh
__global__ void build_gatesT(const float* __restrict__ Wih, const float* __restrict__ Whh,
                             bf16_t* __restrict__ WgT) {
    size_t tid = (size_t)blockIdx.x * blockDim.x + threadIdx.x;
    size_t total = (size_t)NG * KK;
    if (tid >= total) return;
    int n = (int)(tid / KK), k = (int)(tid % KK);
    float v = (k < 2 * DD) ? Wih[(size_t)k * NG + n] : Whh[(size_t)(k - 2 * DD) * NG + n];
    WgT[tid] = f2bf(v);
}

// keys_proj = e_all @ Ua + bu   (M=B*S=65536, K=D, N=D).
// Each wave: 1 m-tile x 4 n-tiles -> 4 WMMAs per A-fragment load.
__global__ void __launch_bounds__(256) keys_gemm(const float* __restrict__ eall,
                          const bf16_t* __restrict__ UaT,
                          const float* __restrict__ bu, float* __restrict__ keys) {
    int wave = blockIdx.x * (blockDim.x >> 5) + (threadIdx.x >> 5);
    int mt = wave >> 3;        // (B*S)/16 = 4096 m-tiles
    int ng = wave & 7;         // 8 groups of 4 n-tiles (64 cols)
    if (mt >= (BB * SS) / 16) return;
    int lane = threadIdx.x & 31, half = lane >> 4, r = lane & 15;
    const float*  arow = eall + (size_t)(mt * 16 + r) * DD;
    const bf16_t* b0 = UaT + (size_t)(ng * 64 + r) * DD;
    const bf16_t* b1 = b0 + (size_t)16 * DD;
    const bf16_t* b2 = b0 + (size_t)32 * DD;
    const bf16_t* b3 = b0 + (size_t)48 * DD;
    v8f a0 = {}, a1 = {}, a2 = {}, a3 = {};
    for (int k0 = 0; k0 < DD; k0 += 32) {
        int ko = k0 + half * 8;
        __builtin_prefetch(arow + ko + 64, 0, 1);   // global_prefetch_b8
        bf16x16 a = load_frag_f32(arow + ko);
        a0 = wmma_bf16(a, load_frag_bf16(b0 + ko), a0);
        a1 = wmma_bf16(a, load_frag_bf16(b1 + ko), a1);
        a2 = wmma_bf16(a, load_frag_bf16(b2 + ko), a2);
        a3 = wmma_bf16(a, load_frag_bf16(b3 + ko), a3);
    }
    float* crow = keys + (size_t)(mt * 16) * DD + ng * 64 + r;
    float c0 = bu[ng * 64 + r], c1 = bu[ng * 64 + 16 + r];
    float c2 = bu[ng * 64 + 32 + r], c3 = bu[ng * 64 + 48 + r];
#pragma unroll
    for (int i = 0; i < 8; ++i) {
        size_t ro = (size_t)(i + half * 8) * DD;
        crow[ro]      = a0[i] + c0;
        crow[ro + 16] = a1[i] + c1;
        crow[ro + 32] = a2[i] + c2;
        crow[ro + 48] = a3[i] + c3;
    }
}

// h=e_h[0], c=e_c[0]; hc_bf16=bf16(h+c); x[:,2D:3D]=bf16(h)
__global__ void init_state(const float* __restrict__ eh, const float* __restrict__ ec,
                           float* __restrict__ h, float* __restrict__ c,
                           bf16_t* __restrict__ hc, bf16_t* __restrict__ x) {
    size_t i = (size_t)blockIdx.x * blockDim.x + threadIdx.x;
    if (i >= (size_t)BB * DD) return;
    float hv = eh[i], cv = ec[i];
    h[i] = hv; c[i] = cv;
    hc[i] = f2bf(hv + cv);
    int b = (int)(i / DD), d = (int)(i % DD);
    x[(size_t)b * KK + 2 * DD + d] = f2bf(hv);
}

// ---------------- per-step kernels ----------------

// q = hc @ Wa + ba   (M=B, K=D, N=D); 1 m-tile x 4 n-tiles per wave
__global__ void __launch_bounds__(256) q_gemm(const bf16_t* __restrict__ hc,
                       const bf16_t* __restrict__ WaT,
                       const float* __restrict__ ba, float* __restrict__ q) {
    int wave = blockIdx.x * (blockDim.x >> 5) + (threadIdx.x >> 5);
    int mt = wave >> 3;        // 32 m-tiles
    int ng = wave & 7;
    if (mt >= BB / 16) return;
    int lane = threadIdx.x & 31, half = lane >> 4, r = lane & 15;
    const bf16_t* arow = hc + (size_t)(mt * 16 + r) * DD;
    const bf16_t* b0 = WaT + (size_t)(ng * 64 + r) * DD;
    const bf16_t* b1 = b0 + (size_t)16 * DD;
    const bf16_t* b2 = b0 + (size_t)32 * DD;
    const bf16_t* b3 = b0 + (size_t)48 * DD;
    v8f a0 = {}, a1 = {}, a2 = {}, a3 = {};
    for (int k0 = 0; k0 < DD; k0 += 32) {
        int ko = k0 + half * 8;
        bf16x16 a = load_frag_bf16(arow + ko);
        a0 = wmma_bf16(a, load_frag_bf16(b0 + ko), a0);
        a1 = wmma_bf16(a, load_frag_bf16(b1 + ko), a1);
        a2 = wmma_bf16(a, load_frag_bf16(b2 + ko), a2);
        a3 = wmma_bf16(a, load_frag_bf16(b3 + ko), a3);
    }
    float* crow = q + (size_t)(mt * 16) * DD + ng * 64 + r;
    float c0 = ba[ng * 64 + r], c1 = ba[ng * 64 + 16 + r];
    float c2 = ba[ng * 64 + 32 + r], c3 = ba[ng * 64 + 48 + r];
#pragma unroll
    for (int i = 0; i < 8; ++i) {
        size_t ro = (size_t)(i + half * 8) * DD;
        crow[ro]      = a0[i] + c0;
        crow[ro + 16] = a1[i] + c1;
        crow[ro + 32] = a2[i] + c2;
        crow[ro + 48] = a3[i] + c3;
    }
}

// One block per batch row: scores -> softmax -> ctx; writes cross-attn and x[:,0:2D] (emb|ctx)
__global__ void attention(const float* __restrict__ q, const float* __restrict__ keys,
                          const float* __restrict__ Va, const float* __restrict__ bv,
                          const float* __restrict__ eall, const float* __restrict__ Eemb,
                          const int* __restrict__ target,
                          float* __restrict__ attn_out, bf16_t* __restrict__ x, int t) {
    int b = blockIdx.x;        // 0..B-1
    int s = threadIdx.x;       // 0..S-1 (128)
    __shared__ float qs[DD];
    __shared__ float vas[DD];
    __shared__ float ws[SS];
    for (int d = s; d < DD; d += SS) { qs[d] = q[(size_t)b * DD + d]; vas[d] = Va[d]; }
    __syncthreads();
    const float* krow = keys + ((size_t)b * SS + s) * DD;
    float sc = bv[0];
    for (int d = 0; d < DD; ++d) sc += tanhf(qs[d] + krow[d]) * vas[d];
    ws[s] = sc;
    __syncthreads();
    float mx = ws[0];
    for (int i = 1; i < SS; ++i) mx = fmaxf(mx, ws[i]);
    float e = expf(sc - mx);
    __syncthreads();
    ws[s] = e;
    __syncthreads();
    float sum = 0.0f;
    for (int i = 0; i < SS; ++i) sum += ws[i];
    float w = e / sum;
    attn_out[((size_t)b * TT + t) * SS + s] = w;
    __syncthreads();
    ws[s] = w;
    __syncthreads();
    int tok = (t == 0) ? 0 : target[(size_t)b * TT + t - 1];
#pragma unroll
    for (int kk = 0; kk < DD / SS; ++kk) {     // 4 columns per thread, coalesced over threads
        int d = s + kk * SS;
        float ctx = 0.0f;
        for (int sp = 0; sp < SS; ++sp) ctx += ws[sp] * eall[((size_t)b * SS + sp) * DD + d];
        x[(size_t)b * KK + DD + d] = f2bf(ctx);
        x[(size_t)b * KK + d]      = f2bf(Eemb[(size_t)tok * DD + d]);
    }
}

// g = x @ [W_ih;W_hh] + biases, fused LSTM cell update (in-place h,c).
// Block = 16 rows (one m-tile), 8 waves = 8 cell-column tiles; each wave does all 4 gates.
// The 16x1536 bf16 A panel (48KB) is staged into LDS by the Tensor Data Mover once per block.
__global__ void __launch_bounds__(256) gates_lstm(const bf16_t* __restrict__ x,
                           const bf16_t* __restrict__ WgT,
                           const float* __restrict__ b_ih, const float* __restrict__ b_hh,
                           float* __restrict__ h, float* __restrict__ c) {
    __shared__ __align__(16) bf16_t Ax[16 * KK];   // 48 KB
    int mt = blockIdx.x >> 2;          // 0..31 m-tile
    int ngrp = blockIdx.x & 3;         // group of 8 n-tiles
    int wid = threadIdx.x >> 5;
    int nt = ngrp * 8 + wid;           // 0..31 cell-column tile
    int lane = threadIdx.x & 31, half = lane >> 4, r = lane & 15;

    // --- TDM: DMA 16 rows x 1536 cols of bf16 x into LDS (wave 0 issues) ---
    if (threadIdx.x < 32) {
        unsigned lds_off = (unsigned)(size_t)&Ax[0];            // addr[31:0] = LDS offset
        unsigned long long ga = (unsigned long long)(size_t)(x + (size_t)(mt * 16) * KK);
        u32x4 g0; i32x8 g1; i32x4 gz4 = {0, 0, 0, 0}; i32x8 gz8 = {0, 0, 0, 0, 0, 0, 0, 0};
        g0[0] = 1u;                                             // count=1, user mode
        g0[1] = lds_off;                                        // lds_addr
        g0[2] = (unsigned)(ga & 0xFFFFFFFFu);                   // global_addr[31:0]
        g0[3] = (unsigned)((ga >> 32) & 0x1FFFFFFu) | (2u << 30); // global_addr[56:32] | type=2
        g1[0] = 0x10000;                                        // wg_mask=0, data_size=1 (2B)
        g1[1] = (int)(((unsigned)KK & 0xFFFFu) << 16);          // tensor_dim0[15:0] @ bits63:48
        g1[2] = (int)((((unsigned)KK >> 16) & 0xFFFFu)          // tensor_dim0[31:16]
                 | (((unsigned)BB & 0xFFFFu) << 16));           // tensor_dim1[15:0]
        g1[3] = (int)(((unsigned)KK & 0xFFFFu) << 16);          // tensor_dim1 hi=0 | tile_dim0
        g1[4] = 16;                                             // tile_dim1=16 rows, tile_dim2=0
        g1[5] = KK;                                             // tensor_dim0_stride[31:0]
        g1[6] = 0;                                              // stride hi, dim1_stride lo
        g1[7] = 0;
        __builtin_amdgcn_tensor_load_to_lds(g0, g1, gz4, gz4, gz8, 0);
        __builtin_amdgcn_s_wait_tensorcnt(0);
    }
    __syncthreads();

    const bf16_t* arow = &Ax[(size_t)r * KK];                   // A frags from LDS
    const bf16_t* b0 = WgT + (size_t)(nt * 16 + r) * KK;        // gate i
    const bf16_t* b1 = b0 + (size_t)DD * KK;                    // gate f
    const bf16_t* b2 = b0 + (size_t)(2 * DD) * KK;              // gate g
    const bf16_t* b3 = b0 + (size_t)(3 * DD) * KK;              // gate o
    v8f a0 = {}, a1 = {}, a2 = {}, a3 = {};
    for (int k0 = 0; k0 < KK; k0 += 32) {
        int ko = k0 + half * 8;
        __builtin_prefetch(b0 + ko + 256, 0, 1);                // global_prefetch_b8
        bf16x16 a = load_frag_bf16(arow + ko);
        a0 = wmma_bf16(a, load_frag_bf16(b0 + ko), a0);
        a1 = wmma_bf16(a, load_frag_bf16(b1 + ko), a1);
        a2 = wmma_bf16(a, load_frag_bf16(b2 + ko), a2);
        a3 = wmma_bf16(a, load_frag_bf16(b3 + ko), a3);
    }
    int n = nt * 16 + r;
    float bi  = b_ih[n]          + b_hh[n];
    float bf_ = b_ih[n + DD]     + b_hh[n + DD];
    float bg  = b_ih[n + 2 * DD] + b_hh[n + 2 * DD];
    float bo  = b_ih[n + 3 * DD] + b_hh[n + 3 * DD];
#pragma unroll
    for (int i = 0; i < 8; ++i) {
        int m = mt * 16 + i + half * 8;
        size_t off = (size_t)m * DD + n;
        float cold = c[off];
        float gi = sigm(a0[i] + bi);
        float gf = sigm(a1[i] + bf_);
        float gg = tanhf(a2[i] + bg);
        float go = sigm(a3[i] + bo);
        float c2 = gf * cold + gi * gg;
        c[off] = c2;
        h[off] = go * tanhf(c2);
    }
}

// Refresh bf16 views of the new state for the next step / output GEMM
__global__ void post_state(const float* __restrict__ h, const float* __restrict__ c,
                           bf16_t* __restrict__ hc, bf16_t* __restrict__ x) {
    size_t i = (size_t)blockIdx.x * blockDim.x + threadIdx.x;
    if (i >= (size_t)BB * DD) return;
    float hv = h[i];
    hc[i] = f2bf(hv + c[i]);
    int b = (int)(i / DD), d = (int)(i % DD);
    x[(size_t)b * KK + 2 * DD + d] = f2bf(hv);
}

// out = h2 @ W_out + b_out, then row log_softmax. Block = 16 rows; 8 waves = 8 col tiles (V=128).
__global__ void __launch_bounds__(256) out_logsoftmax(const bf16_t* __restrict__ x,
                               const bf16_t* __restrict__ WoT,
                               const float* __restrict__ b_out, float* __restrict__ out, int t) {
    __shared__ float tile[16][VV];
    __shared__ float red[16][16];
    __shared__ float rmax[16];
    __shared__ float rsum[16];
    int wid = threadIdx.x >> 5;
    int lane = threadIdx.x & 31, half = lane >> 4, r = lane & 15;
    int m0 = blockIdx.x * 16;
    const bf16_t* arow = x + (size_t)(m0 + r) * KK + 2 * DD;   // h-part of x
    const bf16_t* brow = WoT + (size_t)(wid * 16 + r) * DD;
    v8f acc = {};
    for (int k0 = 0; k0 < DD; k0 += 32) {
        int ko = k0 + half * 8;
        acc = wmma_bf16(load_frag_bf16(arow + ko), load_frag_bf16(brow + ko), acc);
    }
    float bias = b_out[wid * 16 + r];
#pragma unroll
    for (int i = 0; i < 8; ++i) tile[i + half * 8][wid * 16 + r] = acc[i] + bias;
    __syncthreads();
    // log_softmax over 128 cols: 16 rows x 16 partial-reducers (8 cols each)
    int rr = threadIdx.x >> 4, j = threadIdx.x & 15;
    float mx = -3.4e38f;
#pragma unroll
    for (int kk = 0; kk < 8; ++kk) mx = fmaxf(mx, tile[rr][j * 8 + kk]);
    red[rr][j] = mx;
    __syncthreads();
    if (j == 0) {
        float m2 = red[rr][0];
        for (int kk = 1; kk < 16; ++kk) m2 = fmaxf(m2, red[rr][kk]);
        rmax[rr] = m2;
    }
    __syncthreads();
    float rm = rmax[rr];
    float se = 0.0f;
#pragma unroll
    for (int kk = 0; kk < 8; ++kk) se += expf(tile[rr][j * 8 + kk] - rm);
    red[rr][j] = se;
    __syncthreads();
    if (j == 0) {
        float s2 = 0.0f;
        for (int kk = 0; kk < 16; ++kk) s2 += red[rr][kk];
        rsum[rr] = s2;
    }
    __syncthreads();
    float lse = rm + logf(rsum[rr]);
    size_t base = (size_t)(m0 + rr) * TT * VV + (size_t)t * VV;
#pragma unroll
    for (int kk = 0; kk < 8; ++kk) {
        int col = j * 8 + kk;
        out[base + col] = tile[rr][col] - lse;
    }
}

// ---------------- host launcher ----------------

extern "C" void kernel_launch(void* const* d_in, const int* in_sizes, int n_in,
                              void* d_out, int out_size, void* d_ws, size_t ws_size,
                              hipStream_t stream) {
    (void)in_sizes; (void)n_in; (void)out_size; (void)ws_size;
    const float* e_all  = (const float*)d_in[0];
    const float* e_h    = (const float*)d_in[1];
    const float* e_c    = (const float*)d_in[2];
    const int*   target = (const int*)  d_in[3];
    const float* Eemb   = (const float*)d_in[4];
    const float* Wa     = (const float*)d_in[5];
    const float* ba     = (const float*)d_in[6];
    const float* Ua     = (const float*)d_in[7];
    const float* bu     = (const float*)d_in[8];
    const float* Va     = (const float*)d_in[9];
    const float* bv     = (const float*)d_in[10];
    const float* W_ih   = (const float*)d_in[11];
    const float* b_ih   = (const float*)d_in[12];
    const float* W_hh   = (const float*)d_in[13];
    const float* b_hh   = (const float*)d_in[14];
    const float* W_out  = (const float*)d_in[15];
    const float* b_out  = (const float*)d_in[16];

    float* outputs = (float*)d_out;                                // [B,T,V]
    float* attns   = outputs + (size_t)BB * TT * VV;               // [B,T,S]

    // scratch layout (~150 MB)
    char* p = (char*)d_ws;
    auto take = [&](size_t bytes) { char* r = p; p += (bytes + 255) & ~(size_t)255; return r; };
    float*  keys  = (float*) take((size_t)BB * SS * DD * 4);       // 128 MB
    bf16_t* WgT   = (bf16_t*)take((size_t)NG * KK * 2);            // 6 MB
    bf16_t* WaT   = (bf16_t*)take((size_t)DD * DD * 2);
    bf16_t* UaT   = (bf16_t*)take((size_t)DD * DD * 2);
    bf16_t* WoT   = (bf16_t*)take((size_t)VV * DD * 2);
    float*  qbuf  = (float*) take((size_t)BB * DD * 4);
    float*  hbuf  = (float*) take((size_t)BB * DD * 4);
    float*  cbuf  = (float*) take((size_t)BB * DD * 4);
    bf16_t* xbuf  = (bf16_t*)take((size_t)BB * KK * 2);            // [emb|ctx|h] bf16
    bf16_t* hcbuf = (bf16_t*)take((size_t)BB * DD * 2);

    auto blocks = [](size_t total, int bs) { return (unsigned)((total + bs - 1) / bs); };

    // one-time weight prep
    transpose_to_bf16<<<blocks((size_t)DD * DD, 256), 256, 0, stream>>>(Wa, WaT, DD, DD);
    transpose_to_bf16<<<blocks((size_t)DD * DD, 256), 256, 0, stream>>>(Ua, UaT, DD, DD);
    transpose_to_bf16<<<blocks((size_t)DD * VV, 256), 256, 0, stream>>>(W_out, WoT, DD, VV);
    build_gatesT<<<blocks((size_t)NG * KK, 256), 256, 0, stream>>>(W_ih, W_hh, WgT);

    // loop-invariant keys_proj: 4096 m-tiles x 8 n-groups = 32768 waves, 8 waves/block
    keys_gemm<<<4096, 256, 0, stream>>>(e_all, UaT, bu, keys);

    init_state<<<blocks((size_t)BB * DD, 256), 256, 0, stream>>>(e_h, e_c, hbuf, cbuf, hcbuf, xbuf);

    for (int t = 0; t < TT; ++t) {
        q_gemm<<<32, 256, 0, stream>>>(hcbuf, WaT, ba, qbuf);                 // 256 waves
        attention<<<BB, SS, 0, stream>>>(qbuf, keys, Va, bv, e_all, Eemb, target,
                                         attns, xbuf, t);
        gates_lstm<<<128, 256, 0, stream>>>(xbuf, WgT, b_ih, b_hh, hbuf, cbuf);
        post_state<<<blocks((size_t)BB * DD, 256), 256, 0, stream>>>(hbuf, cbuf, hcbuf, xbuf);
        out_logsoftmax<<<32, 256, 0, stream>>>(xbuf, WoT, b_out, outputs, t);
    }
}
